// BaseCPNN_65541200937098
// MI455X (gfx1250) — matrix operand order repaired and checked by reading, not compiled, not run on hardware
//
#include <hip/hip_runtime.h>
#include <hip/hip_bf16.h>

// Problem sizes (fixed by the reference)
#define BSZ 4096
#define DD  1024
#define HH  8192
#define OO  1000

typedef __attribute__((ext_vector_type(16))) __bf16 v16bf;
typedef __attribute__((ext_vector_type(8)))  __bf16 v8bf;
typedef __attribute__((ext_vector_type(4)))  __bf16 v4bf;
typedef __attribute__((ext_vector_type(8)))  float  v8f;

// Types matching the async-copy builtin prototype (from compiler diagnostic:
// param 1 is 'int __vector(4) __device__ *')
typedef int v4i __attribute__((__vector_size__(16)));
typedef __attribute__((address_space(1))) v4i* gv4p;   // global (AS1)
typedef __attribute__((address_space(3))) v4i* lv4p;   // LDS (AS3)

__device__ __forceinline__ unsigned int order_f32(float s) {
  // monotonic float -> u32 map (smaller float => smaller u32)
  unsigned int u = __float_as_uint(s);
  return (u & 0x80000000u) ? ~u : (u | 0x80000000u);
}

__device__ __forceinline__ unsigned long long umin64(unsigned long long a,
                                                     unsigned long long b) {
  return a < b ? a : b;
}

// ---- CDNA5 async global->LDS copy (16B per lane), ASYNCcnt tracked ----
__device__ __forceinline__ void async_cp16(const __bf16* g, const __bf16* l) {
#if __has_builtin(__builtin_amdgcn_global_load_async_to_lds_b128)
  __builtin_amdgcn_global_load_async_to_lds_b128(
      (gv4p)(unsigned long long)(size_t)g,
      (lv4p)(unsigned int)(size_t)l, 0, 0);
#else
  unsigned int lds = (unsigned int)(size_t)l;       // low 32 bits = LDS offset
  unsigned long long ga = (unsigned long long)(size_t)g;
  asm volatile("global_load_async_to_lds_b128 %0, %1, off"
               :: "v"(lds), "v"(ga) : "memory");
#endif
}

template <int N>
__device__ __forceinline__ void wait_async() {
#if __has_builtin(__builtin_amdgcn_s_wait_asynccnt)
  __builtin_amdgcn_s_wait_asynccnt((unsigned short)N);
#else
  asm volatile("s_wait_asynccnt %0" :: "n"(N) : "memory");
#endif
}

__global__ void init_argmin(unsigned long long* __restrict__ packed) {
  int i = blockIdx.x * blockDim.x + threadIdx.x;
  if (i < BSZ) packed[i] = 0xFFFFFFFFFFFFFFFFULL;
}

// w2[h] = sum_d W[h,d]^2  (one 256-thread block per row)
__global__ void row_sumsq(const float* __restrict__ W, float* __restrict__ w2) {
  int h = blockIdx.x;
  int tid = threadIdx.x;
  const float4* row = (const float4*)(W + (size_t)h * DD);
  float4 v = row[tid];
  float s = v.x * v.x + v.y * v.y + v.z * v.z + v.w * v.w;
  #pragma unroll
  for (int m = 16; m >= 1; m >>= 1) s += __shfl_xor(s, m, 32);
  __shared__ float wsum[8];
  if ((tid & 31) == 0) wsum[tid >> 5] = s;
  __syncthreads();
  if (tid == 0) {
    float t = 0.f;
    #pragma unroll
    for (int i = 0; i < 8; ++i) t += wsum[i];
    w2[h] = t;
  }
}

// f32 -> bf16 bulk conversion (4 elems/thread)
__global__ void f32_to_bf16(const float* __restrict__ in, __bf16* __restrict__ out,
                            int n4) {
  int i = blockIdx.x * blockDim.x + threadIdx.x;
  if (i < n4) {
    float4 v = ((const float4*)in)[i];
    v4bf b = { (__bf16)v.x, (__bf16)v.y, (__bf16)v.z, (__bf16)v.w };
    ((v4bf*)out)[i] = b;
  }
}

// =====================================================================
// Fast path: bf16 inputs, async global->LDS double-buffered, 4x4 WMMA/wave
// Block tile 128(M) x 256(N) x 64(K); 8 waves, each wave 64x64.
// =====================================================================
#define BM2 128
#define BN2 256
#define BK2 64
#define LDW 72   // row stride in bf16 (144B: 16B aligned, bank-skewed)
#define NCOPIES 12  // async b128 issued per wave per K-chunk (A:4 + B:8)

__global__ __launch_bounds__(256)
void gemm_argmin_async(const __bf16* __restrict__ Xb, const __bf16* __restrict__ Wb,
                       const float* __restrict__ w2,
                       unsigned long long* __restrict__ packed) {
  __shared__ __bf16 sA[2][BM2 * LDW];
  __shared__ __bf16 sB[2][BN2 * LDW];
  __shared__ unsigned long long rowMin[BM2];

  const int tid  = threadIdx.x;
  const int lane = tid & 31;
  const int wave = tid >> 5;
  const int wm   = wave & 1;     // 2 waves along M (64 rows each)
  const int wn   = wave >> 1;    // 4 waves along N (64 cols each)
  const int hf   = lane >> 4;
  const int lm   = lane & 15;

  const int mBase = blockIdx.y * BM2;
  const int nBase = blockIdx.x * BN2;

  if (tid < BM2) rowMin[tid] = 0xFFFFFFFFFFFFFFFFULL;

  v8f acc[4][4] = {};

  auto issue = [&](int kc, int buf) {
    const int k0 = kc * BK2;
    // A tile: 128 rows x 64 bf16 = 1024 x 16B segments
    #pragma unroll
    for (int p = 0; p < 4; ++p) {
      int s = tid + 256 * p;
      int row = s >> 3, c16 = s & 7;
      async_cp16(Xb + (size_t)(mBase + row) * DD + k0 + c16 * 8,
                 &sA[buf][row * LDW + c16 * 8]);
    }
    // B tile: 256 rows x 64 bf16 = 2048 x 16B segments
    #pragma unroll
    for (int p = 0; p < 8; ++p) {
      int s = tid + 256 * p;
      int row = s >> 3, c16 = s & 7;
      async_cp16(Wb + (size_t)(nBase + row) * DD + k0 + c16 * 8,
                 &sB[buf][row * LDW + c16 * 8]);
    }
  };

  const int nk = DD / BK2;   // 16
  issue(0, 0);

  for (int kc = 0; kc < nk; ++kc) {
    const int buf = kc & 1;
    if (kc + 1 < nk) {
      issue(kc + 1, buf ^ 1);
      wait_async<NCOPIES>();   // previous chunk's copies (in-order) done
    } else {
      wait_async<0>();
    }
    __syncthreads();           // all waves' copies visible

    #pragma unroll
    for (int ks = 0; ks < BK2; ks += 32) {
      v16bf aF[4], bF[4];
      #pragma unroll
      for (int tm = 0; tm < 4; ++tm) {
        int m = wm * 64 + tm * 16 + lm;
        const __bf16* p0 = &sA[buf][m * LDW + ks + hf * 8];
        ((v8bf*)&aF[tm])[0] = *(const v8bf*)(p0);        // ds_load_b128
        ((v8bf*)&aF[tm])[1] = *(const v8bf*)(p0 + 16);
      }
      #pragma unroll
      for (int tn = 0; tn < 4; ++tn) {
        int n = wn * 64 + tn * 16 + lm;
        const __bf16* p0 = &sB[buf][n * LDW + ks + hf * 8];
        ((v8bf*)&bF[tn])[0] = *(const v8bf*)(p0);
        ((v8bf*)&bF[tn])[1] = *(const v8bf*)(p0 + 16);
      }
      #pragma unroll
      for (int tm = 0; tm < 4; ++tm)
        #pragma unroll
        for (int tn = 0; tn < 4; ++tn)
          acc[tm][tn] = __builtin_amdgcn_wmma_f32_16x16x32_bf16(
              false, aF[tm], false, bF[tn], (short)0, acc[tm][tn], false, false);
    }
    __syncthreads();           // done reading buf before it is refilled
  }

  // Epilogue: per-row argmin of score = w2[h] - 2*dot
  float w2v[4];
  #pragma unroll
  for (int tn = 0; tn < 4; ++tn)
    w2v[tn] = w2[nBase + wn * 64 + tn * 16 + lm];

  #pragma unroll
  for (int tm = 0; tm < 4; ++tm) {
    #pragma unroll
    for (int r = 0; r < 8; ++r) {
      int localRow = wm * 64 + tm * 16 + hf * 8 + r;
      unsigned long long best = 0xFFFFFFFFFFFFFFFFULL;
      #pragma unroll
      for (int tn = 0; tn < 4; ++tn) {
        int h = nBase + wn * 64 + tn * 16 + lm;
        float score = w2v[tn] - 2.0f * acc[tm][tn][r];
        unsigned long long key =
            ((unsigned long long)order_f32(score) << 32) | (unsigned int)h;
        best = umin64(best, key);
      }
      #pragma unroll
      for (int m = 1; m <= 8; m <<= 1)
        best = umin64(best, __shfl_xor(best, m, 32));
      if (lm == 0) atomicMin(&rowMin[localRow], best);     // ds_min_u64
    }
  }
  __syncthreads();
  if (tid < BM2) atomicMin(&packed[mBase + tid], rowMin[tid]); // global_atomic_min_u64
}

// =====================================================================
// Fallback path (small ws): f32 loads + in-kernel bf16 conversion
// =====================================================================
#define BM 64
#define BN 128
#define BK 64
#define LDA 72
#define LDB 72

__global__ __launch_bounds__(256)
void gemm_argmin(const float* __restrict__ X, const float* __restrict__ W,
                 const float* __restrict__ w2,
                 unsigned long long* __restrict__ packed) {
  __shared__ __bf16 sA[BM * LDA];
  __shared__ __bf16 sB[BN * LDB];
  __shared__ unsigned long long rowMin[BM];

  const int tid  = threadIdx.x;
  const int lane = tid & 31;
  const int wave = tid >> 5;
  const int wm   = wave & 1;
  const int wn   = wave >> 1;
  const int hf   = lane >> 4;
  const int lm   = lane & 15;

  const int mBase = blockIdx.y * BM;
  const int nBase = blockIdx.x * BN;

  if (tid < BM) rowMin[tid] = 0xFFFFFFFFFFFFFFFFULL;

  v8f acc[2][2] = {};

  const int rowL = tid >> 4;
  const int colL = (tid & 15) * 4;

  for (int k0 = 0; k0 < DD; k0 += BK) {
    __syncthreads();
    #pragma unroll
    for (int p = 0; p < 4; ++p) {
      int r = p * 16 + rowL;
      const float4 v = *(const float4*)(X + (size_t)(mBase + r) * DD + k0 + colL);
      v4bf b = { (__bf16)v.x, (__bf16)v.y, (__bf16)v.z, (__bf16)v.w };
      *(v4bf*)(&sA[r * LDA + colL]) = b;
    }
    #pragma unroll
    for (int p = 0; p < 8; ++p) {
      int r = p * 16 + rowL;
      const float4 v = *(const float4*)(W + (size_t)(nBase + r) * DD + k0 + colL);
      v4bf b = { (__bf16)v.x, (__bf16)v.y, (__bf16)v.z, (__bf16)v.w };
      *(v4bf*)(&sB[r * LDB + colL]) = b;
    }
    __syncthreads();

    #pragma unroll
    for (int ks = 0; ks < BK; ks += 32) {
      v16bf aF[2], bF[2];
      #pragma unroll
      for (int tm = 0; tm < 2; ++tm) {
        int m = wm * 32 + tm * 16 + lm;
        const __bf16* p0 = &sA[m * LDA + ks + hf * 8];
        ((v8bf*)&aF[tm])[0] = *(const v8bf*)(p0);
        ((v8bf*)&aF[tm])[1] = *(const v8bf*)(p0 + 16);
      }
      #pragma unroll
      for (int tn = 0; tn < 2; ++tn) {
        int n = wn * 32 + tn * 16 + lm;
        const __bf16* p0 = &sB[n * LDB + ks + hf * 8];
        ((v8bf*)&bF[tn])[0] = *(const v8bf*)(p0);
        ((v8bf*)&bF[tn])[1] = *(const v8bf*)(p0 + 16);
      }
      #pragma unroll
      for (int tm = 0; tm < 2; ++tm)
        #pragma unroll
        for (int tn = 0; tn < 2; ++tn)
          acc[tm][tn] = __builtin_amdgcn_wmma_f32_16x16x32_bf16(
              false, aF[tm], false, bF[tn], (short)0, acc[tm][tn], false, false);
    }
  }

  float w2v[2];
  #pragma unroll
  for (int tn = 0; tn < 2; ++tn)
    w2v[tn] = w2[nBase + wn * 32 + tn * 16 + lm];

  #pragma unroll
  for (int tm = 0; tm < 2; ++tm) {
    #pragma unroll
    for (int r = 0; r < 8; ++r) {
      int localRow = wm * 32 + tm * 16 + hf * 8 + r;
      unsigned long long best = 0xFFFFFFFFFFFFFFFFULL;
      #pragma unroll
      for (int tn = 0; tn < 2; ++tn) {
        int h = nBase + wn * 32 + tn * 16 + lm;
        float score = w2v[tn] - 2.0f * acc[tm][tn][r];
        unsigned long long key =
            ((unsigned long long)order_f32(score) << 32) | (unsigned int)h;
        best = umin64(best, key);
      }
      #pragma unroll
      for (int m = 1; m <= 8; m <<= 1)
        best = umin64(best, __shfl_xor(best, m, 32));
      if (lm == 0) atomicMin(&rowMin[localRow], best);
    }
  }
  __syncthreads();
  if (tid < BM) atomicMin(&packed[mBase + tid], rowMin[tid]);
}

// out[b,o] = G[o, winner_b]; winners (as float) appended after B*O elements
__global__ void gather_out(const float* __restrict__ G,
                           const unsigned long long* __restrict__ packed,
                           float* __restrict__ out) {
  int b = blockIdx.x;
  int o = blockIdx.y * 256 + threadIdx.x;
  unsigned int w = (unsigned int)(packed[b] & 0xFFFFFFFFULL);
  if (o < OO) out[(size_t)b * OO + o] = G[(size_t)o * HH + w];
  if (o == 0) out[(size_t)BSZ * OO + b] = (float)w;
}

extern "C" void kernel_launch(void* const* d_in, const int* in_sizes, int n_in,
                              void* d_out, int out_size, void* d_ws, size_t ws_size,
                              hipStream_t stream) {
  const float* X = (const float*)d_in[0];  // [B, D]
  const float* W = (const float*)d_in[1];  // [H, D]
  const float* G = (const float*)d_in[2];  // [O, H]
  float* out = (float*)d_out;

  float* w2 = (float*)d_ws;                                              // 32 KB
  unsigned long long* packed =
      (unsigned long long*)((char*)d_ws + (size_t)HH * sizeof(float));   // 32 KB

  const size_t off_xb = 65536;
  const size_t off_wb = off_xb + (size_t)BSZ * DD * 2;                   // +8 MB
  const size_t need   = off_wb + (size_t)HH * DD * 2;                    // +16 MB

  init_argmin<<<(BSZ + 255) / 256, 256, 0, stream>>>(packed);
  row_sumsq<<<HH, 256, 0, stream>>>(W, w2);

  if (ws_size >= need) {
    __bf16* Xb = (__bf16*)((char*)d_ws + off_xb);
    __bf16* Wb = (__bf16*)((char*)d_ws + off_wb);
    f32_to_bf16<<<(BSZ * DD / 4 + 255) / 256, 256, 0, stream>>>(X, Xb, BSZ * DD / 4);
    f32_to_bf16<<<(HH * DD / 4 + 255) / 256, 256, 0, stream>>>(W, Wb, HH * DD / 4);
    dim3 grid(HH / BN2, BSZ / BM2);   // (32, 32)
    gemm_argmin_async<<<grid, 256, 0, stream>>>(Xb, Wb, w2, packed);
  } else {
    dim3 grid(HH / BN, BSZ / BM);     // (64, 64)
    gemm_argmin<<<grid, 256, 0, stream>>>(X, W, w2, packed);
  }

  gather_out<<<dim3(BSZ, (OO + 255) / 256), 256, 0, stream>>>(G, packed, out);
}
